// QMultiHeadedAttention_36498632082244
// MI455X (gfx1250) — matrix operand order repaired and checked by reading, not compiled, not run on hardware
//
#include <hip/hip_runtime.h>
#include <hip/hip_bf16.h>

// ---------------------------------------------------------------------------
// Problem constants (B, W, D, H) = (4, 1024, 1024, 16), HS = 64
// ---------------------------------------------------------------------------
#define SEQ    1024
#define DMODEL 1024
#define NHEAD  16
#define HSZ    64
// softmax done in exp2 domain: v_exp_f32 computes 2^x natively
#define SCALE_L2E 0.1803368801111244f   /* (1/sqrt(64)) * log2(e) */
#define NEG_FILL_L2 (-1.0e9f)

typedef __attribute__((ext_vector_type(16))) __bf16   v16bf;
typedef __attribute__((ext_vector_type(8)))  float    v8f;
typedef __attribute__((ext_vector_type(4)))  unsigned uint4v;

union Frag { uint4v u[2]; v16bf v; };

// native casts -> backend picks the hardware bf16 convert (packed where possible)
__device__ __forceinline__ unsigned short bf16b(float f) {
  union { __bf16 h; unsigned short u; } x; x.h = (__bf16)f; return x.u;
}
__device__ __forceinline__ unsigned pack2(float lo, float hi) {
  union { __bf16 h[2]; unsigned u; } x;
  x.h[0] = (__bf16)lo; x.h[1] = (__bf16)hi; return x.u;
}
__device__ __forceinline__ v8f wmma_bf16(const Frag& a, const Frag& b, v8f c) {
  return __builtin_amdgcn_wmma_f32_16x16x32_bf16(false, a.v, false, b.v,
                                                 (short)0, c, false, false);
}
// reductions across the 16 lanes (one half-wave) that co-own a C-matrix row
__device__ __forceinline__ float red16_max(float v) {
#pragma unroll
  for (int m = 8; m >= 1; m >>= 1) v = fmaxf(v, __shfl_xor(v, m, 32));
  return v;
}
__device__ __forceinline__ float red16_sum(float v) {
#pragma unroll
  for (int m = 8; m >= 1; m >>= 1) v += __shfl_xor(v, m, 32);
  return v;
}

// ---------------------------------------------------------------------------
// GEMM: out(4096x1024) = A(4096x1024) @ Wt(1024x1024) + bias
//   A_BF16  : A is bf16 (else f32, converted to bf16 while staging to LDS)
//   OUT_MODE: 0 = bf16 row-major   (q/v projections)
//             1 = bf16 per-head transposed kT[b][h][kw][d]  (k projection)
//             2 = f32 row-major    (final output)
// 128x128 block tile, 4 waves of 64x64, K-step 32, all math in v_wmma bf16.
// ---------------------------------------------------------------------------
template <bool A_BF16, int OUT_MODE>
__global__ __launch_bounds__(128) void gemm_kernel(
    const void* __restrict__ Aptr, const float* __restrict__ Wt,
    const float* __restrict__ bias, void* __restrict__ outp) {
  constexpr int N = DMODEL, K = DMODEL;
  constexpr int AS = 40, BS = 40;           // LDS row strides (bf16 elems, +8 pad)
  __shared__ unsigned short As[128 * AS];   // A tile, row-major [m][k]
  __shared__ unsigned short Bs[128 * BS];   // W tile, transposed [n][k]

  const int tid  = threadIdx.x;
  const int lane = tid & 31;
  const int wave = tid >> 5;
  const int hv   = (lane >> 4) & 1;         // wave half (lanes 0-15 / 16-31)
  const int l16  = lane & 15;
  const int waveM = (wave >> 1) * 64;
  const int waveN = (wave & 1) * 64;
  const int tileM = blockIdx.x * 128;
  const int tileN = blockIdx.y * 128;

  v8f acc[4][4];
#pragma unroll
  for (int i = 0; i < 4; ++i)
#pragma unroll
    for (int j = 0; j < 4; ++j) acc[i][j] = (v8f){};

  float bvals[4];
#pragma unroll
  for (int nt = 0; nt < 4; ++nt)
    bvals[nt] = bias[tileN + waveN + nt * 16 + l16];

  const int kpair = tid & 15;               // k-row pair for W transpose
  const int nbase = (tid >> 4) * 16;        // 16-column group

  for (int k0 = 0; k0 < K; k0 += 32) {
    // ---- stage A tile (thread t owns row t; 32 k-values) ----
    if (A_BF16) {
      const unsigned short* src =
          (const unsigned short*)Aptr + (size_t)(tileM + tid) * K + k0;
      uint4v* dst = (uint4v*)&As[tid * AS];
#pragma unroll
      for (int i = 0; i < 4; ++i) dst[i] = ((const uint4v*)src)[i];
    } else {
      const float* src = (const float*)Aptr + (size_t)(tileM + tid) * K + k0;
      uint4v* dst = (uint4v*)&As[tid * AS];
#pragma unroll
      for (int i = 0; i < 4; ++i) {
        float4 a = ((const float4*)src)[2 * i];
        float4 c = ((const float4*)src)[2 * i + 1];
        uint4v u;
        u[0] = pack2(a.x, a.y); u[1] = pack2(a.z, a.w);
        u[2] = pack2(c.x, c.y); u[3] = pack2(c.z, c.w);
        dst[i] = u;
      }
    }
    // ---- stage W tile transposed: Bs[n][k] = W[k0+k][tileN+n] ----
    {
      const float* w0 = Wt + (size_t)(k0 + 2 * kpair) * N + tileN + nbase;
      const float* w1 = w0 + N;
#pragma unroll
      for (int i = 0; i < 4; ++i) {
        float4 a = ((const float4*)w0)[i];
        float4 c = ((const float4*)w1)[i];
        *(unsigned*)&Bs[(nbase + 4 * i + 0) * BS + 2 * kpair] = pack2(a.x, c.x);
        *(unsigned*)&Bs[(nbase + 4 * i + 1) * BS + 2 * kpair] = pack2(a.y, c.y);
        *(unsigned*)&Bs[(nbase + 4 * i + 2) * BS + 2 * kpair] = pack2(a.z, c.z);
        *(unsigned*)&Bs[(nbase + 4 * i + 3) * BS + 2 * kpair] = pack2(a.w, c.w);
      }
    }
    __syncthreads();

    // ---- fragments + 16 WMMAs ----
    Frag af[4], bfm[4];
#pragma unroll
    for (int mt = 0; mt < 4; ++mt) {
      const unsigned short* base = &As[(waveM + mt * 16 + l16) * AS];
      af[mt].u[0] = *(const uint4v*)(base + hv * 8);        // k = hv*8 + e
      af[mt].u[1] = *(const uint4v*)(base + 16 + hv * 8);   // k = 16 + hv*8 + e
    }
#pragma unroll
    for (int nt = 0; nt < 4; ++nt) {
      const unsigned short* base = &Bs[(waveN + nt * 16 + l16) * BS];
      bfm[nt].u[0] = *(const uint4v*)(base + hv * 8);       // k = hv*8 + e
      bfm[nt].u[1] = *(const uint4v*)(base + 16 + hv * 8);
    }
#pragma unroll
    for (int mt = 0; mt < 4; ++mt)
#pragma unroll
      for (int nt = 0; nt < 4; ++nt)
        acc[mt][nt] = wmma_bf16(af[mt], bfm[nt], acc[mt][nt]);
    __syncthreads();
  }

  // ---- epilogue: C row = r + 8*hv, col = l16 ----
#pragma unroll
  for (int mt = 0; mt < 4; ++mt) {
#pragma unroll
    for (int nt = 0; nt < 4; ++nt) {
      const int col = tileN + waveN + nt * 16 + l16;
#pragma unroll
      for (int r = 0; r < 8; ++r) {
        const int m = tileM + waveM + mt * 16 + hv * 8 + r;
        const float v = acc[mt][nt][r] + bvals[nt];
        if constexpr (OUT_MODE == 2) {
          ((float*)outp)[(size_t)m * N + col] = v;
        } else if constexpr (OUT_MODE == 0) {
          ((unsigned short*)outp)[(size_t)m * N + col] = bf16b(v);
        } else {  // kT[b][h][kw=col][d]: m = b*1024 + h*64 + d
          const int bb = m >> 10, r1 = m & 1023, hh = r1 >> 6, dd = r1 & 63;
          ((unsigned short*)outp)[(((size_t)(bb * 16 + hh) * 1024 + col) << 6) +
                                  dd] = bf16b(v);
        }
      }
    }
  }
}

// ---------------------------------------------------------------------------
// Flash attention per (q-block of 128, head, batch). 4 waves x 32 query rows.
// Q fragments resident in registers; K B-fragments straight from the
// pre-transposed kT layout in global; V transposed into LDS; online softmax
// in registers (exp2 domain) via 16-lane shuffles; P staged in wave-private
// LDS as A-fragments for the P@V WMMA. Score block processed one 16-row
// m-tile at a time to keep peak VGPRs under 256 (no vgpr-MSB remapping).
// Reference index maps baked in:
//   Qh/Vh[w,d] = proj[b, h*64 + w/16, (w%16)*64 + d]
//   Kh[d,kw]   = k_proj[b, h*64 + d, kw]  ==  kT[b][h][kw][d]
// ---------------------------------------------------------------------------
__global__ __launch_bounds__(128) void attn_kernel(
    const unsigned short* __restrict__ qp, const unsigned short* __restrict__ kT,
    const unsigned short* __restrict__ vp, const unsigned char* __restrict__ mask,
    unsigned short* __restrict__ ctx) {
  __shared__ unsigned short Vt[64 * 136];    // V^T tile: [d][w], 17.0 KB
  __shared__ unsigned short Ps[128 * 136];   // P (bf16): [qrow][kw], 34.0 KB

  const int tid  = threadIdx.x;
  const int lane = tid & 31;
  const int wave = tid >> 5;
  const int hv   = (lane >> 4) & 1;
  const int l16  = lane & 15;
  const int qb = blockIdx.x, h = blockIdx.y, b = blockIdx.z;

  // ---- Q fragments, resident for the whole kernel ----
  Frag qa[2][2];
#pragma unroll
  for (int mt = 0; mt < 2; ++mt) {
    const int qrow = qb * 128 + wave * 32 + mt * 16 + l16;
    const unsigned short* qbase =
        qp + (size_t)(b * SEQ + h * HSZ + (qrow >> 4)) * DMODEL +
        (qrow & 15) * 64;
#pragma unroll
    for (int di = 0; di < 2; ++di) {
      qa[mt][di].u[0] = *(const uint4v*)(qbase + di * 32 + hv * 8);
      qa[mt][di].u[1] = *(const uint4v*)(qbase + di * 32 + 16 + hv * 8);
    }
  }

  v8f o[2][4];
#pragma unroll
  for (int mt = 0; mt < 2; ++mt)
#pragma unroll
    for (int nv = 0; nv < 4; ++nv) o[mt][nv] = (v8f){};
  float runMax[2][8], runSum[2][8];
#pragma unroll
  for (int mt = 0; mt < 2; ++mt)
#pragma unroll
    for (int r = 0; r < 8; ++r) { runMax[mt][r] = -3.0e38f; runSum[mt][r] = 0.f; }

  const size_t kheadBase = (size_t)(b * 16 + h) * 1024;  // row base into kT
  const int d  = tid & 63;                               // V-transpose lane role
  const int wg = (tid >> 6) * 64;

  for (int j = 0; j < 8; ++j) {
    const int kv0 = j * 128;
    __syncthreads();  // Vt about to be overwritten; prior P@V must be done

    // ---- cooperative transpose of V block into LDS ----
    // kv0+wg is a multiple of 64, so the gather offsets fold to unroll-time
    // constants off one hoisted base pointer (immediate-offset loads).
    {
      const unsigned short* vbase =
          vp + (size_t)(b * SEQ + h * HSZ + ((kv0 + wg) >> 4)) * DMODEL + d;
#pragma unroll 8
      for (int w2 = 0; w2 < 64; w2 += 2) {
        unsigned short e0 = vbase[((w2 + 0) >> 4) * DMODEL + ((w2 + 0) & 15) * 64];
        unsigned short e1 = vbase[((w2 + 1) >> 4) * DMODEL + ((w2 + 1) & 15) * 64];
        *(unsigned*)&Vt[d * 136 + wg + w2] = (unsigned)e0 | ((unsigned)e1 << 16);
      }
    }
    if (j < 7)  // prefetch next K block (global_prefetch_b8)
      __builtin_prefetch(kT + ((kheadBase + kv0 + 128 + tid) << 6), 0, 2);
    __syncthreads();

    bool mvals[8];
#pragma unroll
    for (int nt = 0; nt < 8; ++nt)
      mvals[nt] = mask[(size_t)b * SEQ + kv0 + nt * 16 + l16] != 0;

    // ---- per 16-row m-tile: S = Q@K, masked online softmax, P store ----
#pragma unroll
    for (int mt = 0; mt < 2; ++mt) {
      v8f s[8];
#pragma unroll
      for (int nt = 0; nt < 8; ++nt) s[nt] = (v8f){};
#pragma unroll
      for (int nt = 0; nt < 8; ++nt) {
        Frag kb[2];
        const unsigned short* kbase =
            kT + ((kheadBase + kv0 + nt * 16 + l16) << 6);
#pragma unroll
        for (int di = 0; di < 2; ++di) {
          kb[di].u[0] = *(const uint4v*)(kbase + di * 32 + hv * 8);
          kb[di].u[1] = *(const uint4v*)(kbase + di * 32 + 16 + hv * 8);
        }
#pragma unroll
        for (int di = 0; di < 2; ++di)
          s[nt] = wmma_bf16(qa[mt][di], kb[di], s[nt]);
      }
#pragma unroll
      for (int r = 0; r < 8; ++r) {
        float sv[8];
        float m = -3.0e38f;
#pragma unroll
        for (int nt = 0; nt < 8; ++nt) {  // exp2-domain scores
          float x = mvals[nt] ? s[nt][r] * SCALE_L2E : NEG_FILL_L2;
          sv[nt] = x;
          m = fmaxf(m, x);
        }
        m = red16_max(m);                 // row max across 16 lanes
        const float nm = fmaxf(runMax[mt][r], m);
        const float corr = __builtin_exp2f(runMax[mt][r] - nm);
        runMax[mt][r] = nm;
        float rs = 0.f;
        const int prow = wave * 32 + mt * 16 + hv * 8 + r;
#pragma unroll
        for (int nt = 0; nt < 8; ++nt) {
          const float p = __builtin_exp2f(sv[nt] - nm);
          rs += p;
          Ps[prow * 136 + nt * 16 + l16] = bf16b(p);
        }
        rs = red16_sum(rs);
        runSum[mt][r] = runSum[mt][r] * corr + rs;
#pragma unroll
        for (int nv = 0; nv < 4; ++nv)    // rescale O row in place
          o[mt][nv][r] *= corr;
      }
    }

    // ---- O += P @ V : 32 WMMAs per wave (wave-private Ps, no barrier) ----
#pragma unroll
    for (int k0i = 0; k0i < 4; ++k0i) {
      const int k0 = k0i * 32;
      Frag pa[2];
#pragma unroll
      for (int mt = 0; mt < 2; ++mt) {
        const unsigned short* base = &Ps[(wave * 32 + mt * 16 + l16) * 136];
        pa[mt].u[0] = *(const uint4v*)(base + k0 + hv * 8);
        pa[mt].u[1] = *(const uint4v*)(base + k0 + 16 + hv * 8);
      }
#pragma unroll
      for (int nv = 0; nv < 4; ++nv) {
        Frag vb;
        const unsigned short* base = &Vt[(nv * 16 + l16) * 136 + k0 + hv * 8];
        vb.u[0] = *(const uint4v*)(base);
        vb.u[1] = *(const uint4v*)(base + 16);
#pragma unroll
        for (int mt = 0; mt < 2; ++mt)
          o[mt][nv] = wmma_bf16(pa[mt], vb, o[mt][nv]);
      }
    }
  }

  // ---- finalize: O /= l, store ctx[b][w][h*64+d] (bf16) ----
#pragma unroll
  for (int mt = 0; mt < 2; ++mt) {
#pragma unroll
    for (int r = 0; r < 8; ++r) {
      const float inv = 1.0f / runSum[mt][r];
      const int qrow = qb * 128 + wave * 32 + mt * 16 + hv * 8 + r;
#pragma unroll
      for (int nv = 0; nv < 4; ++nv) {
        const int col = h * HSZ + nv * 16 + l16;
        ctx[(size_t)(b * SEQ + qrow) * DMODEL + col] = bf16b(o[mt][nv][r] * inv);
      }
    }
  }
}

// ---------------------------------------------------------------------------
extern "C" void kernel_launch(void* const* d_in, const int* in_sizes, int n_in,
                              void* d_out, int out_size, void* d_ws,
                              size_t ws_size, hipStream_t stream) {
  (void)in_sizes; (void)n_in; (void)out_size; (void)ws_size;
  const float* k_in = (const float*)d_in[0];
  const float* v_in = (const float*)d_in[1];
  const float* q_in = (const float*)d_in[2];
  const unsigned char* mask = (const unsigned char*)d_in[3];
  const float* Wk = (const float*)d_in[4];
  const float* bk = (const float*)d_in[5];
  const float* Wv = (const float*)d_in[6];
  const float* bv = (const float*)d_in[7];
  const float* Wq = (const float*)d_in[8];
  const float* bq = (const float*)d_in[9];
  const float* Wo = (const float*)d_in[10];
  const float* bo = (const float*)d_in[11];

  // workspace: 4 bf16 buffers of 4096x1024 (8 MB each, 32 MB total)
  unsigned short* qp  = (unsigned short*)d_ws;
  unsigned short* kTp = qp  + (size_t)4096 * 1024;
  unsigned short* vpp = kTp + (size_t)4096 * 1024;
  unsigned short* ctx = vpp + (size_t)4096 * 1024;

  dim3 gg(32, 8, 1);  // 4096/128 x 1024/128
  gemm_kernel<false, 0><<<gg, 128, 0, stream>>>(q_in, Wq, bq, qp);
  gemm_kernel<false, 1><<<gg, 128, 0, stream>>>(k_in, Wk, bk, kTp);
  gemm_kernel<false, 0><<<gg, 128, 0, stream>>>(v_in, Wv, bv, vpp);
  attn_kernel<<<dim3(8, NHEAD, 4), 128, 0, stream>>>(qp, kTp, vpp, mask, ctx);
  gemm_kernel<true, 2><<<gg, 128, 0, stream>>>(ctx, Wo, bo, d_out);
}
